// MultiHeadSelfAttention_54142357734082
// MI455X (gfx1250) — matrix operand order repaired and checked by reading, compiled, not run on hardware
//
#include <hip/hip_runtime.h>

typedef __attribute__((ext_vector_type(16))) _Float16 v16h;
typedef __attribute__((ext_vector_type(8)))  float    v8f;
typedef __attribute__((ext_vector_type(4)))  unsigned int v4u;
typedef __attribute__((ext_vector_type(8)))  int      v8i;
typedef __attribute__((ext_vector_type(4)))  int      v4i;

#define WMMA(A, B, C) \
  __builtin_amdgcn_wmma_f32_16x16x32_f16(false, (A), false, (B), (short)0, (C), false, false)

#if defined(__has_builtin)
#if __has_builtin(__builtin_amdgcn_tensor_load_to_lds) && \
    __has_builtin(__builtin_amdgcn_s_wait_tensorcnt)
#define USE_TDM 1
#endif
#endif
#ifndef USE_TDM
#define USE_TDM 0
#endif

namespace {
constexpr int   kDim   = 1024;
constexpr int   kHeads = 16;
constexpr int   kHdim  = 64;
constexpr int   kSeq   = 2048;
constexpr int   kBatch = 2;
constexpr int   kRows  = kBatch * kSeq;  // 4096
constexpr float kScale = 0.125f;         // 1/sqrt(64)
}  // namespace

// ---- WMMA fragment loaders (CDNA5 16-bit layouts, CDNA5 ISA §7.12.2) ----
// A (16x32, MxK): lane half h=lane>>4; row m=lane&15. Elements 0..7 hold K in
// [8h, 8h+8), elements 8..15 hold K in [16+8h, 16+8h+8)  -> two contiguous runs.
__device__ inline v16h load_afrag(const _Float16* rowp, int k0, int half) {
  v16h a;
  const _Float16* p = rowp + k0 + 8 * half;
#pragma unroll
  for (int i = 0; i < 8; ++i) {
    a[i]     = p[i];
    a[i + 8] = p[i + 16];
  }
  return a;
}

// B (32x16, KxN) from a tile stored [N][K]: lane khalf=lane>>4, col n=lane&15;
// elements 0..7 hold K in [8*khalf, 8*khalf+8), 8..15 hold K in [16+8*khalf, +8).
__device__ inline v16h load_bfrag(const _Float16* colp, int k0, int khalf) {
  v16h b;
  const _Float16* p = colp + k0 + 8 * khalf;
#pragma unroll
  for (int i = 0; i < 8; ++i) {
    b[i]     = p[i];
    b[i + 8] = p[i + 16];
  }
  return b;
}

__device__ inline v8f vzero8() {
  v8f z = {0.f, 0.f, 0.f, 0.f, 0.f, 0.f, 0.f, 0.f};
  return z;
}

#if USE_TDM
// Byte offset of a __shared__ object within the workgroup LDS allocation.
__device__ inline unsigned lds_byte_off(const void* p) {
  return (unsigned)(unsigned long long)(const __attribute__((address_space(3))) char*)p;
}

// TDM: async DMA of a rows x cols f16 tile (row stride `src_stride` elems) from
// global memory into LDS at byte offset `lds_addr`, with LDS row padding of
// 4 DWORDs after every 32 DWORDs (128B row -> 144B padded row = stride 72 f16).
__device__ inline void tdm_load_tile_64x64_pad(unsigned lds_addr,
                                               const _Float16* gsrc) {
  unsigned long long ga = (unsigned long long)(const void*)gsrc;
  v4u g0;
  g0[0] = 1u;                                        // count=1
  g0[1] = lds_addr;                                  // lds_addr (bytes)
  g0[2] = (unsigned)(ga & 0xffffffffu);              // global_addr[31:0]
  g0[3] = (unsigned)((ga >> 32) & 0x01ffffffu) | (2u << 30);  // addr[56:32],type=2
  v8i g1;
  g1[0] = (int)((1u << 16) |        // data_size = 2 bytes
                (1u << 20) |        // pad_enable
                (4u << 22) |        // pad_interval: every 32 DWORDs
                (3u << 25));        // pad_amount: 4 DWORDs
  g1[1] = (int)(64u << 16);         // tensor_dim0 = 64
  g1[2] = (int)(64u << 16);         // tensor_dim1 = 64
  g1[3] = (int)(64u << 16);         // tile_dim0 = 64
  g1[4] = 64;                       // tile_dim1 = 64 (tile_dim2 = 0)
  g1[5] = 64;                       // tensor_dim0_stride = 64
  g1[6] = 0;
  g1[7] = 0;
  v4i gz = {0, 0, 0, 0};
#if __clang_major__ >= 23
  v8i gz8 = {0, 0, 0, 0, 0, 0, 0, 0};
  __builtin_amdgcn_tensor_load_to_lds(g0, g1, gz, gz, gz8, 0);
#else
  __builtin_amdgcn_tensor_load_to_lds(g0, g1, gz, gz, 0);
#endif
}
#endif  // USE_TDM

// ================= Kernel 1: QKV projection =================
// C[4096x3072] = X[4096x1024] @ Wqkv[1024x3072]; scatter into Q/K/V [B,H,N,hd] f16.
// WG tile 64x256; 8 waves as 2(M) x 4(N); wave tile 32x64 (2x4 WMMA accumulators).
__global__ __launch_bounds__(256) void qkv_gemm_kernel(
    const float* __restrict__ x, const float* __restrict__ wqkv,
    _Float16* __restrict__ qkv_ws) {
  __shared__ _Float16 lds_a[64 * 40];    // X tile 64x32, stride 40
  __shared__ _Float16 lds_bt[256 * 40];  // W tile stored [N=256][K=32], stride 40

  const int t    = threadIdx.x;
  const int lane = t & 31;
  const int w    = t >> 5;
  const int half = lane >> 4;
  const int n15  = lane & 15;
  const int wm   = (w >> 2) * 32;  // wave M offset (0/32)
  const int wn   = (w & 3) * 64;   // wave N offset (0..192)
  const int row0 = blockIdx.x * 64;
  const int col0 = blockIdx.y * 256;

  v8f acc[2][4];
#pragma unroll
  for (int i = 0; i < 2; ++i)
#pragma unroll
    for (int j = 0; j < 4; ++j) acc[i][j] = vzero8();

  for (int kk = 0; kk < kDim; kk += 32) {
    __syncthreads();
#pragma unroll
    for (int it = 0; it < 8; ++it) {  // X: 64x32 f32 -> f16
      int i = it * 256 + t;
      int r = i >> 5, c = i & 31;
      lds_a[r * 40 + c] = (_Float16)x[(size_t)(row0 + r) * kDim + kk + c];
    }
#pragma unroll
    for (int it = 0; it < 32; ++it) {  // W: 32x256 f32 -> f16, transposed to [N][K]
      int i = it * 256 + t;
      int kr = i >> 8, c = i & 255;
      lds_bt[c * 40 + kr] = (_Float16)wqkv[(size_t)(kk + kr) * (3 * kDim) + col0 + c];
    }
    __syncthreads();

    v16h a0 = load_afrag(&lds_a[(wm + n15) * 40], 0, half);
    v16h a1 = load_afrag(&lds_a[(wm + 16 + n15) * 40], 0, half);
#pragma unroll
    for (int j = 0; j < 4; ++j) {
      v16h b = load_bfrag(&lds_bt[(wn + j * 16 + n15) * 40], 0, half);
      acc[0][j] = WMMA(a0, b, acc[0][j]);
      acc[1][j] = WMMA(a1, b, acc[1][j]);
    }
  }

#pragma unroll
  for (int i = 0; i < 2; ++i)
#pragma unroll
    for (int j = 0; j < 4; ++j)
#pragma unroll
      for (int r = 0; r < 8; ++r) {
        int R = row0 + wm + i * 16 + r + 8 * half;
        int C = col0 + wn + j * 16 + n15;
        int which = C >> 10;  // 0=q,1=k,2=v
        int d = C & 1023;
        int h = d >> 6, hd = d & 63;
        int b = R >> 11, n = R & 2047;
        qkv_ws[(size_t)which * kRows * kDim +
               ((size_t)((b * kHeads + h) * kSeq + n)) * kHdim + hd] =
            (_Float16)acc[i][j][r];
      }
}

// ================= Kernel 2: flash attention =================
// Grid: (B*H, SEQ/128). WG = 128 query rows; each of 8 waves owns 16 rows.
__global__ __launch_bounds__(256) void flash_attn_kernel(
    const _Float16* __restrict__ q, const _Float16* __restrict__ k,
    const _Float16* __restrict__ v, _Float16* __restrict__ ctx) {
  __shared__ _Float16 lds_k[64 * 72];      // K block [key][hd], stride 72
  __shared__ _Float16 lds_vt[64 * 72];     // V block transposed [hd][key]
  __shared__ _Float16 lds_p[8 * 16 * 72];  // per-wave P strip [16][64]

  const int t    = threadIdx.x;
  const int lane = t & 31;
  const int w    = t >> 5;
  const int half = lane >> 4;
  const int n15  = lane & 15;
  const int bh   = blockIdx.x;  // b*16 + h
  const int rb   = blockIdx.y;  // row block of 128
  const int b    = bh >> 4;
  const int h    = bh & 15;

  // Resident Q A-fragments for this wave's 16-row strip.
  const _Float16* qrow =
      q + ((size_t)bh * kSeq + rb * 128 + w * 16 + n15) * kHdim;
  v16h aq0 = load_afrag(qrow, 0, half);
  v16h aq1 = load_afrag(qrow, 32, half);

  float m_i[8], l_i[8];
  v8f o[4];
#pragma unroll
  for (int r = 0; r < 8; ++r) { m_i[r] = -1e30f; l_i[r] = 0.f; }
#pragma unroll
  for (int ti = 0; ti < 4; ++ti) o[ti] = vzero8();

  const size_t kvbase = (size_t)bh * kSeq * kHdim;
#if USE_TDM
  const unsigned lds_k_off = lds_byte_off(&lds_k[0]);
#endif

  for (int kb = 0; kb < kSeq / 64; ++kb) {
    __syncthreads();
#if USE_TDM
    // Async DMA of the K block via the Tensor Data Mover; the TDM applies the
    // 72-element padded row layout itself (pad fields in D# group1).
    if (w == 0) {
      tdm_load_tile_64x64_pad(lds_k_off, k + kvbase + (size_t)(kb * 64) * kHdim);
    }
#else
#pragma unroll
    for (int it = 0; it < 16; ++it) {  // K block: 64x64 f16, row-major copy
      int i = it * 256 + t;
      int key = i >> 6, hd = i & 63;
      lds_k[key * 72 + hd] = k[kvbase + (size_t)(kb * 64 + key) * kHdim + hd];
    }
#endif
#pragma unroll
    for (int it = 0; it < 16; ++it) {  // V block: transpose to [hd][key]
      int i = it * 256 + t;
      int key = i >> 6, hd = i & 63;
      lds_vt[hd * 72 + key] = v[kvbase + (size_t)(kb * 64 + key) * kHdim + hd];
    }
#if USE_TDM
    if (w == 0) __builtin_amdgcn_s_wait_tensorcnt(0);
#endif
    __syncthreads();

    // S = Q @ K^T : 4 n-tiles x 2 k-steps
    v8f s[4];
#pragma unroll
    for (int j = 0; j < 4; ++j) {
      const _Float16* kr = &lds_k[(j * 16 + n15) * 72];
      v8f a = vzero8();
      a = WMMA(aq0, load_bfrag(kr, 0, half), a);
      a = WMMA(aq1, load_bfrag(kr, 32, half), a);
      s[j] = a;
    }
#pragma unroll
    for (int j = 0; j < 4; ++j)
#pragma unroll
      for (int r = 0; r < 8; ++r) s[j][r] *= kScale;

    // Online softmax. Rows live across 16 lanes {same half}; butterfly reduce.
#pragma unroll
    for (int r = 0; r < 8; ++r) {
      float mx = fmaxf(fmaxf(s[0][r], s[1][r]), fmaxf(s[2][r], s[3][r]));
#pragma unroll
      for (int m = 1; m <= 8; m <<= 1) mx = fmaxf(mx, __shfl_xor(mx, m, 32));
      float mnew  = fmaxf(m_i[r], mx);
      float alpha = __expf(m_i[r] - mnew);
      m_i[r] = mnew;
      float rs = 0.f;
#pragma unroll
      for (int j = 0; j < 4; ++j) {
        float pv = __expf(s[j][r] - mnew);
        s[j][r] = pv;
        rs += pv;
      }
#pragma unroll
      for (int m = 1; m <= 8; m <<= 1) rs += __shfl_xor(rs, m, 32);
      l_i[r] = l_i[r] * alpha + rs;
#pragma unroll
      for (int ti = 0; ti < 4; ++ti) o[ti][r] *= alpha;
    }

    // P: C-layout regs -> per-wave LDS strip [16][64] (A-layout source)
    _Float16* pw = &lds_p[w * 16 * 72];
#pragma unroll
    for (int j = 0; j < 4; ++j)
#pragma unroll
      for (int r = 0; r < 8; ++r)
        pw[(r + 8 * half) * 72 + j * 16 + n15] = (_Float16)s[j][r];
    __syncthreads();

    // O += P @ V
    const _Float16* prow = pw + n15 * 72;
    v16h ap0 = load_afrag(prow, 0, half);
    v16h ap1 = load_afrag(prow, 32, half);
#pragma unroll
    for (int ti = 0; ti < 4; ++ti) {
      const _Float16* vr = &lds_vt[(ti * 16 + n15) * 72];
      o[ti] = WMMA(ap0, load_bfrag(vr, 0, half), o[ti]);
      o[ti] = WMMA(ap1, load_bfrag(vr, 32, half), o[ti]);
    }
  }

  // Normalize and write context [B,N,DIM] as f16.
#pragma unroll
  for (int r = 0; r < 8; ++r) {
    float inv = 1.0f / l_i[r];
    int nrow = rb * 128 + w * 16 + r + 8 * half;
#pragma unroll
    for (int ti = 0; ti < 4; ++ti)
      ctx[((size_t)(b * kSeq) + nrow) * kDim + h * kHdim + ti * 16 + n15] =
          (_Float16)(o[ti][r] * inv);
  }
}

// ================= Kernel 3: output projection + bias =================
// WG tile 64x256; wave tile 32x64.
__global__ __launch_bounds__(256) void proj_gemm_kernel(
    const _Float16* __restrict__ ctx, const float* __restrict__ wproj,
    const float* __restrict__ bias, float* __restrict__ out) {
  __shared__ _Float16 lds_a[64 * 40];
  __shared__ _Float16 lds_bt[256 * 40];

  const int t    = threadIdx.x;
  const int lane = t & 31;
  const int w    = t >> 5;
  const int half = lane >> 4;
  const int n15  = lane & 15;
  const int wm   = (w >> 2) * 32;
  const int wn   = (w & 3) * 64;
  const int row0 = blockIdx.x * 64;
  const int col0 = blockIdx.y * 256;

  v8f acc[2][4];
#pragma unroll
  for (int i = 0; i < 2; ++i)
#pragma unroll
    for (int j = 0; j < 4; ++j) acc[i][j] = vzero8();

  for (int kk = 0; kk < kDim; kk += 32) {
    __syncthreads();
#pragma unroll
    for (int it = 0; it < 8; ++it) {
      int i = it * 256 + t;
      int r = i >> 5, c = i & 31;
      lds_a[r * 40 + c] = ctx[(size_t)(row0 + r) * kDim + kk + c];
    }
#pragma unroll
    for (int it = 0; it < 32; ++it) {
      int i = it * 256 + t;
      int kr = i >> 8, c = i & 255;
      lds_bt[c * 40 + kr] = (_Float16)wproj[(size_t)(kk + kr) * kDim + col0 + c];
    }
    __syncthreads();

    v16h a0 = load_afrag(&lds_a[(wm + n15) * 40], 0, half);
    v16h a1 = load_afrag(&lds_a[(wm + 16 + n15) * 40], 0, half);
#pragma unroll
    for (int j = 0; j < 4; ++j) {
      v16h b = load_bfrag(&lds_bt[(wn + j * 16 + n15) * 40], 0, half);
      acc[0][j] = WMMA(a0, b, acc[0][j]);
      acc[1][j] = WMMA(a1, b, acc[1][j]);
    }
  }

#pragma unroll
  for (int i = 0; i < 2; ++i)
#pragma unroll
    for (int j = 0; j < 4; ++j)
#pragma unroll
      for (int r = 0; r < 8; ++r) {
        int R = row0 + wm + i * 16 + r + 8 * half;
        int C = col0 + wn + j * 16 + n15;
        out[(size_t)R * kDim + C] = acc[i][j][r] + bias[C];
      }
}

extern "C" void kernel_launch(void* const* d_in, const int* in_sizes, int n_in,
                              void* d_out, int out_size, void* d_ws, size_t ws_size,
                              hipStream_t stream) {
  (void)in_sizes; (void)n_in; (void)out_size; (void)ws_size;
  const float* x      = (const float*)d_in[0];
  const float* w_qkv  = (const float*)d_in[1];
  const float* w_proj = (const float*)d_in[2];
  const float* b_proj = (const float*)d_in[3];
  float* out = (float*)d_out;

  _Float16* ws  = (_Float16*)d_ws;
  _Float16* qW  = ws;                             // [B,H,N,hd] x3 (f16)
  _Float16* kW  = ws + (size_t)kRows * kDim;
  _Float16* vW  = ws + (size_t)2 * kRows * kDim;
  _Float16* ctx = ws + (size_t)3 * kRows * kDim;  // [B,N,DIM] (f16)

  qkv_gemm_kernel<<<dim3(kRows / 64, (3 * kDim) / 256), 256, 0, stream>>>(
      x, w_qkv, qW);
  flash_attn_kernel<<<dim3(kBatch * kHeads, kSeq / 128), 256, 0, stream>>>(
      qW, kW, vW, ctx);
  proj_gemm_kernel<<<dim3(kRows / 64, kDim / 256), 256, 0, stream>>>(
      ctx, w_proj, b_proj, out);
}
